// AttentionQMixer_14585708937766
// MI455X (gfx1250) — matrix operand order repaired and checked by reading, not compiled, not run on hardware
//
#include <hip/hip_runtime.h>
#include <hip/hip_bf16.h>

typedef _Float16 h16;
typedef __attribute__((ext_vector_type(16))) _Float16 v16h;
typedef __attribute__((ext_vector_type(8)))  float    v8f;

// ---------------------------------------------------------------------------
// Generic fused GEMM:  out[M,N] = epi( A[M,K] @ W[N,K]^T + bias )
// Tiles: BM=128, BN=64, BK=32. 256 threads = 8 waves; wave w computes rows
// [w*16, w*16+16) x 64 cols via 4 WMMA accumulators. Output stored as f16.
// EPI: 0=none, 1=relu, 2=abs, 3=obs-encoder epilogue (+act gather + q embed)
// ---------------------------------------------------------------------------
template<int EPI, bool AHALF>
__global__ __launch_bounds__(256)
void gemm_wmma_kernel(const void* __restrict__ Ain, const float* __restrict__ W,
                      const float* __restrict__ bias, h16* __restrict__ out,
                      int M, int N, int K,
                      const int* __restrict__ act, const float* __restrict__ qsv,
                      const float* __restrict__ Wact, const float* __restrict__ bact,
                      const float* __restrict__ Wq, const float* __restrict__ bq)
{
    constexpr int APAD = 40; // halves per LDS row (32 + 8 pad) -> 80B, 16B aligned
    __shared__ __align__(16) h16 As[128 * APAD];
    __shared__ __align__(16) h16 Ws[64 * APAD];

    const int tid  = threadIdx.x;
    const int wave = tid >> 5;
    const int lane = tid & 31;
    const int m0   = blockIdx.x * 128;
    const int n0   = blockIdx.y * 64;

    v8f c[4];
    #pragma unroll
    for (int t = 0; t < 4; ++t) c[t] = (v8f){0.f,0.f,0.f,0.f,0.f,0.f,0.f,0.f};

    for (int k0 = 0; k0 < K; k0 += 32) {
        // ---- load A tile (128 rows x 32 K) into LDS as f16 ----
        if (AHALF) {
            // 128 rows x 32 halves = 512 uint4 (8 halves each) -> 2 per thread
            const h16* A = (const h16*)Ain;
            #pragma unroll
            for (int i = 0; i < 2; ++i) {
                int id = tid + i * 256;
                int row = id >> 2, seg = id & 3;
                const uint4 v = *((const uint4*)(A + (size_t)(m0 + row) * K + k0 + seg * 8));
                *((uint4*)(As + row * APAD + seg * 8)) = v;
            }
            // prefetch next K tile (global_prefetch_b8)
            if (k0 + 32 < K) {
                int row = tid >> 1;
                __builtin_prefetch(A + (size_t)(m0 + row) * K + k0 + 32, 0, 1);
            }
        } else {
            // 128 rows x 32 floats = 1024 float4 -> 4 per thread
            const float* A = (const float*)Ain;
            #pragma unroll
            for (int i = 0; i < 4; ++i) {
                int id = tid + i * 256;
                int row = id >> 3, c4 = id & 7;
                float4 f = *((const float4*)(A + (size_t)(m0 + row) * K + k0 + c4 * 4));
                h16* d = As + row * APAD + c4 * 4;
                d[0] = (h16)f.x; d[1] = (h16)f.y; d[2] = (h16)f.z; d[3] = (h16)f.w;
            }
            if (k0 + 32 < K) {
                int row = tid >> 1;
                __builtin_prefetch(A + (size_t)(m0 + row) * K + k0 + 32 + (tid & 1) * 16, 0, 1);
            }
        }
        // ---- load W tile (64 rows x 32) into LDS as f16 (B^T = W rows) ----
        #pragma unroll
        for (int i = 0; i < 2; ++i) {
            int id = tid + i * 256;
            int row = id >> 3, c4 = id & 7;
            float4 f = *((const float4*)(W + (size_t)(n0 + row) * K + k0 + c4 * 4));
            h16* d = Ws + row * APAD + c4 * 4;
            d[0] = (h16)f.x; d[1] = (h16)f.y; d[2] = (h16)f.z; d[3] = (h16)f.w;
        }
        __syncthreads();

        // ---- A fragment per ISA 16-bit 16x32 layout ----
        union { v16h v; uint4 u[2]; } af;
        {
            int mrow = wave * 16 + (lane & 15);
            int off  = (lane < 16) ? 0 : 8;
            const h16* p = As + mrow * APAD;
            af.u[0] = *((const uint4*)(p + off));
            af.u[1] = *((const uint4*)(p + 16 + off));
        }
        #pragma unroll
        for (int t = 0; t < 4; ++t) {
            union { v16h v; uint4 u[2]; } bf;
            int nrow = t * 16 + (lane & 15);
            int koff = (lane < 16) ? 0 : 8;
            const h16* p = Ws + nrow * APAD + koff;
            bf.u[0] = *((const uint4*)(p));
            bf.u[1] = *((const uint4*)(p + 16));
            c[t] = __builtin_amdgcn_wmma_f32_16x16x32_f16(
                       false, af.v, false, bf.v, (short)0, c[t], false, false);
        }
        __syncthreads();
    }

    // ---- epilogue + f16 store (C layout: vgpr j -> M=j or j+8; N=lane%16) ----
    #pragma unroll
    for (int j = 0; j < 8; ++j) {
        int mrow = wave * 16 + j + ((lane < 16) ? 0 : 8);
        size_t gm = (size_t)(m0 + mrow);
        int   ra = 0; float rq = 0.f;
        if (EPI == 3) { ra = act[gm]; rq = qsv[gm]; }
        #pragma unroll
        for (int t = 0; t < 4; ++t) {
            int n = n0 + t * 16 + (lane & 15);
            float val = c[t][j];
            if (EPI == 3) {
                val += bias[n] + bact[n] + Wact[n * 32 + ra] + rq * Wq[n] + bq[n];
            } else {
                val += bias[n];
                if (EPI == 1) val = fmaxf(val, 0.f);
                else if (EPI == 2) val = fabsf(val);
            }
            out[gm * (size_t)N + n] = (h16)val;
        }
    }
}

// ---------------------------------------------------------------------------
// Fused QKV GEMM + 8-agent multi-head attention + folded (Wo,Wa2q) projection.
// Block = 128 rows = 16 agent-groups. qkv (128x192) never leaves LDS.
// ---------------------------------------------------------------------------
__global__ __launch_bounds__(256)
void qkv_attn_kernel(const h16* __restrict__ x, const float* __restrict__ Wqkv,
                     const float* __restrict__ bqkv, const float* __restrict__ wc,
                     float* __restrict__ aq)
{
    constexpr int PAD  = 72;  // halves; 144B rows
    constexpr int QPAD = 200; // halves; 400B rows
    __shared__ __align__(16) char smem[128 * QPAD * 2]; // 51200B, As/Ws overlap qkv
    h16* As = (h16*)smem;                   // 128 x 72 = 18432B
    h16* Ws = (h16*)(smem + 128 * PAD * 2); // 192 x 72 = 27648B (ends 46080)
    h16* Qk = (h16*)smem;                   // reused after barrier: 128 x 200

    const int tid = threadIdx.x, wave = tid >> 5, lane = tid & 31;
    const size_t m0 = (size_t)blockIdx.x * 128;

    // load x tile: 128 rows x 64 halves = 1024 uint4 (8 halves each) -> 4/thread
    #pragma unroll
    for (int i = 0; i < 4; ++i) {
        int id = tid + i * 256;
        int row = id >> 3, seg = id & 7;
        *((uint4*)(As + row * PAD + seg * 8)) =
            *((const uint4*)(x + (m0 + row) * 64 + seg * 8));
    }
    // load Wqkv (192 x 64 f32) -> f16 LDS
    #pragma unroll
    for (int i = 0; i < 12; ++i) {
        int id = tid + i * 256;
        int row = id >> 4, c4 = id & 15;
        float4 f = *((const float4*)(Wqkv + row * 64 + c4 * 4));
        h16* d = Ws + row * PAD + c4 * 4;
        d[0] = (h16)f.x; d[1] = (h16)f.y; d[2] = (h16)f.z; d[3] = (h16)f.w;
    }
    __syncthreads();

    v8f c[12];
    #pragma unroll
    for (int t = 0; t < 12; ++t) c[t] = (v8f){0.f,0.f,0.f,0.f,0.f,0.f,0.f,0.f};

    #pragma unroll
    for (int ks = 0; ks < 2; ++ks) {
        int k0 = ks * 32;
        union { v16h v; uint4 u[2]; } af;
        int mrow = wave * 16 + (lane & 15);
        int off  = (lane < 16) ? 0 : 8;
        const h16* p = As + mrow * PAD + k0;
        af.u[0] = *((const uint4*)(p + off));
        af.u[1] = *((const uint4*)(p + 16 + off));
        #pragma unroll
        for (int t = 0; t < 12; ++t) {
            union { v16h v; uint4 u[2]; } bf;
            int n = t * 16 + (lane & 15);
            int koff = (lane < 16) ? 0 : 8;
            const h16* q = Ws + n * PAD + k0 + koff;
            bf.u[0] = *((const uint4*)q);
            bf.u[1] = *((const uint4*)(q + 16));
            c[t] = __builtin_amdgcn_wmma_f32_16x16x32_f16(
                       false, af.v, false, bf.v, (short)0, c[t], false, false);
        }
    }
    __syncthreads(); // everyone done reading As/Ws before overlay

    // write qkv tile (+bias) into LDS
    #pragma unroll
    for (int j = 0; j < 8; ++j) {
        int mrow = wave * 16 + j + ((lane < 16) ? 0 : 8);
        #pragma unroll
        for (int t = 0; t < 12; ++t) {
            int n = t * 16 + (lane & 15);
            Qk[mrow * QPAD + n] = (h16)(c[t][j] + bqkv[n]);
        }
    }
    __syncthreads();

    // attention: lane = a*4 + h (A=8 rows, H=4 heads, HD=16); wave -> 2 groups
    const float c0 = wc[64];
    const int a = lane >> 2, h = lane & 3;
    #pragma unroll
    for (int gi = 0; gi < 2; ++gi) {
        int g = wave * 2 + gi;
        const h16* base = Qk + (g * 8) * QPAD;
        float qreg[16];
        #pragma unroll
        for (int d = 0; d < 16; ++d) qreg[d] = (float)base[a * QPAD + h * 16 + d];
        float s[8], mx = -1e30f;
        #pragma unroll
        for (int b = 0; b < 8; ++b) {
            const h16* kp = base + b * QPAD + 64 + h * 16;
            float acc = 0.f;
            #pragma unroll
            for (int d = 0; d < 16; ++d) acc += qreg[d] * (float)kp[d];
            s[b] = acc * 0.25f; // 1/sqrt(HD=16)
            mx = fmaxf(mx, s[b]);
        }
        float num = 0.f, den = 0.f;
        #pragma unroll
        for (int b = 0; b < 8; ++b) {
            float e = __expf(s[b] - mx);
            den += e;
            const h16* vp = base + b * QPAD + 128 + h * 16;
            float vd = 0.f;
            #pragma unroll
            for (int d = 0; d < 16; ++d) vd += (float)vp[d] * wc[h * 16 + d];
            num += e * vd;
        }
        float part = num / den;
        part += __shfl_xor(part, 1, 32);
        part += __shfl_xor(part, 2, 32);
        if (h == 0) aq[m0 + g * 8 + a] = part + c0;
    }
}

// w_comb[e] = sum_f Wa2q[f]*Wo[f,e]; wc[64] = Wa2q.bo + ba2q
__global__ void setup_wcomb_kernel(const float* __restrict__ Wo, const float* __restrict__ bo,
                                   const float* __restrict__ Wa2q, const float* __restrict__ ba2q,
                                   float* __restrict__ wc)
{
    int e = threadIdx.x;
    float s = 0.f;
    for (int f = 0; f < 64; ++f) s += Wa2q[f] * Wo[f * 64 + e];
    wc[e] = s;
    if (e == 0) {
        float t = 0.f;
        for (int f = 0; f < 64; ++f) t += Wa2q[f] * bo[f];
        wc[64] = t + ba2q[0];
    }
}

// Final mixing: one wave per sample n. hidden = elu(sum_a aq*w1 + b1);
// y = hidden.wf + relu(v1).Wv2 + bv2
__global__ __launch_bounds__(256)
void final_mix_kernel(const float* __restrict__ aq, const h16* __restrict__ w1,
                      const h16* __restrict__ b1, const h16* __restrict__ wf,
                      const h16* __restrict__ v1, const float* __restrict__ Wv2,
                      const float* __restrict__ bv2, float* __restrict__ out)
{
    int wave = threadIdx.x >> 5, lane = threadIdx.x & 31;
    size_t n = (size_t)blockIdx.x * 8 + wave;
    float aqv[8];
    #pragma unroll
    for (int a = 0; a < 8; ++a) aqv[a] = aq[n * 8 + a];
    float tot = 0.f;
    #pragma unroll
    for (int halfE = 0; halfE < 2; ++halfE) {
        int e = lane + halfE * 32;
        float acc = (float)b1[n * 64 + e];
        #pragma unroll
        for (int a = 0; a < 8; ++a) acc += aqv[a] * (float)w1[n * 512 + a * 64 + e];
        float hid = acc > 0.f ? acc : (__expf(acc) - 1.f);
        tot += hid * (float)wf[n * 64 + e];
        tot += (float)v1[n * 64 + e] * Wv2[e]; // v1 stored post-relu
    }
    #pragma unroll
    for (int m = 16; m >= 1; m >>= 1) tot += __shfl_xor(tot, m, 32);
    if (lane == 0) out[n] = tot + bv2[0];
}

// ---------------------------------------------------------------------------
extern "C" void kernel_launch(void* const* d_in, const int* in_sizes, int n_in,
                              void* d_out, int out_size, void* d_ws, size_t ws_size,
                              hipStream_t stream)
{
    (void)in_sizes; (void)n_in; (void)out_size; (void)ws_size;
    const float* agent_qs     = (const float*)d_in[0];
    const float* states       = (const float*)d_in[1];
    const float* observations = (const float*)d_in[2];
    const int*   actions      = (const int*)  d_in[3];
    const float* W_obs = (const float*)d_in[4];  const float* b_obs = (const float*)d_in[5];
    const float* W_act = (const float*)d_in[6];  const float* b_act = (const float*)d_in[7];
    const float* W_q   = (const float*)d_in[8];  const float* b_q   = (const float*)d_in[9];
    const float* Wqkv  = (const float*)d_in[10]; const float* bqkv  = (const float*)d_in[11];
    const float* Wo    = (const float*)d_in[12]; const float* bo    = (const float*)d_in[13];
    const float* Wa2q  = (const float*)d_in[14]; const float* ba2q  = (const float*)d_in[15];
    const float* Wh1a  = (const float*)d_in[16]; const float* bh1a  = (const float*)d_in[17];
    const float* Wh1b  = (const float*)d_in[18]; const float* bh1b  = (const float*)d_in[19];
    const float* Whfa  = (const float*)d_in[20]; const float* bhfa  = (const float*)d_in[21];
    const float* Whfb  = (const float*)d_in[22]; const float* bhfb  = (const float*)d_in[23];
    const float* Wb1   = (const float*)d_in[24]; const float* bb1   = (const float*)d_in[25];
    const float* Wv1   = (const float*)d_in[26]; const float* bv1   = (const float*)d_in[27];
    const float* Wv2   = (const float*)d_in[28]; const float* bv2   = (const float*)d_in[29];

    const int N  = 64 * 256;      // 16384 samples
    const int MA = N * 8;         // 131072 agent rows

    char* ws = (char*)d_ws;
    // workspace layout (w1 overlays dead x): total ~40.4 MB
    h16*   x_h  = (h16*)(ws + 0);                 // [131072,64]  16.78MB (dead after qkv)
    h16*   w1_h = (h16*)(ws + 0);                 // [16384,512]  16.78MB (overlay)
    h16*   h1a  = (h16*)(ws + 16777216);          // [16384,256]   8.39MB
    h16*   hf   = (h16*)(ws + 25165824);          // [16384,256]   8.39MB
    h16*   v1   = (h16*)(ws + 33554432);          // [16384,64]    2.10MB
    h16*   b1   = (h16*)(ws + 35651584);          // [16384,64]    2.10MB
    h16*   wfh  = (h16*)(ws + 37748736);          // [16384,64]    2.10MB
    float* aq   = (float*)(ws + 39845888);        // [131072]      0.52MB
    float* wc   = (float*)(ws + 40370176);        // [65]
    float* out  = (float*)d_out;

    setup_wcomb_kernel<<<1, 64, 0, stream>>>(Wo, bo, Wa2q, ba2q, wc);

    // x = obs@W_obs^T + b_obs + W_act[.,act] + b_act + qs*W_q + b_q   [131072,64] f16
    gemm_wmma_kernel<3, false><<<dim3(MA / 128, 1), 256, 0, stream>>>(
        observations, W_obs, b_obs, x_h, MA, 64, 256,
        actions, agent_qs, W_act, b_act, W_q, b_q);

    // fused qkv + attention + folded output projection -> attended_qs [131072]
    qkv_attn_kernel<<<MA / 128, 256, 0, stream>>>(x_h, Wqkv, bqkv, wc, aq);

    // hypernet stage 1 over states [16384,512]
    gemm_wmma_kernel<1, false><<<dim3(N / 128, 4), 256, 0, stream>>>(
        states, Wh1a, bh1a, h1a, N, 256, 512, nullptr, nullptr, nullptr, nullptr, nullptr, nullptr);
    gemm_wmma_kernel<1, false><<<dim3(N / 128, 4), 256, 0, stream>>>(
        states, Whfa, bhfa, hf, N, 256, 512, nullptr, nullptr, nullptr, nullptr, nullptr, nullptr);
    gemm_wmma_kernel<1, false><<<dim3(N / 128, 1), 256, 0, stream>>>(
        states, Wv1, bv1, v1, N, 64, 512, nullptr, nullptr, nullptr, nullptr, nullptr, nullptr);
    gemm_wmma_kernel<0, false><<<dim3(N / 128, 1), 256, 0, stream>>>(
        states, Wb1, bb1, b1, N, 64, 512, nullptr, nullptr, nullptr, nullptr, nullptr, nullptr);

    // hypernet stage 2 (f16 inputs)
    gemm_wmma_kernel<2, true><<<dim3(N / 128, 8), 256, 0, stream>>>(
        h1a, Wh1b, bh1b, w1_h, N, 512, 256, nullptr, nullptr, nullptr, nullptr, nullptr, nullptr);
    gemm_wmma_kernel<2, true><<<dim3(N / 128, 1), 256, 0, stream>>>(
        hf, Whfb, bhfb, wfh, N, 64, 256, nullptr, nullptr, nullptr, nullptr, nullptr, nullptr);

    // final mix -> y [16384]
    final_mix_kernel<<<N / 8, 256, 0, stream>>>(aq, w1_h, b1, wfh, v1, Wv2, bv2, out);
}
